// binary_classification_model_56324201120092
// MI455X (gfx1250) — matrix operand order repaired and checked by reading, not compiled, not run on hardware
//
#include <hip/hip_runtime.h>
#include <hip/hip_bf16.h>

#define BB 16
#define SS 1024
#define EMBD 1024
#define PPD 512
#define KW 9
#define HEADS 8
#define LDIM 64   // PPD / HEADS

typedef __attribute__((ext_vector_type(16))) __bf16 v16bf;
typedef __attribute__((ext_vector_type(8)))  float  v8f;

// ---------- bf16 helpers (round-to-nearest-even) ----------
__device__ __forceinline__ __bf16 f2bf(float f) {
  unsigned u = __builtin_bit_cast(unsigned, f);
  u += 0x7FFFu + ((u >> 16) & 1u);
  unsigned short s = (unsigned short)(u >> 16);
  return __builtin_bit_cast(__bf16, s);
}
__device__ __forceinline__ __bf16 bf_zero() {
  unsigned short z = 0;
  return __builtin_bit_cast(__bf16, z);
}

// ---------- WMMA fragment gathers (CDNA5 16-bit layouts, ISA 7.12.2) ----------
// A 16x32 (MxK): lanes0-15 -> M=lane, V0-3:K=0..7, V4-7:K=16..23; lanes16-31: K+=8
__device__ __forceinline__ int a_kidx(int v, int e, int h) {
  return (v < 4) ? (v * 2 + e + h * 8) : (16 + (v - 4) * 2 + e + h * 8);
}
// A operand, memory is row-major [M][K], leading dim ld (elements).
// Per-lane K runs are 8 contiguous elements -> compiler emits ds_load_b128 pairs.
__device__ __forceinline__ v16bf load_a_frag(const __bf16* src, int ld) {
  int lane = threadIdx.x & 31;
  int m = lane & 15, h = lane >> 4;
  v16bf f;
#pragma unroll
  for (int v = 0; v < 8; ++v)
#pragma unroll
    for (int e = 0; e < 2; ++e)
      f[v * 2 + e] = src[m * ld + a_kidx(v, e, h)];
  return f;
}
// B 32x16 (KxN): lane holds column n=lane&15; same K order as A: V0-3:K=0..7, V4-7:K=16..23; lanes16-31: K+=8.
// Memory is [N][K] (transposed staging) so each lane's 16 K-elements form two contiguous runs of 8
// -> two ds_load_b128 per fragment instead of 16 ds_load_u16.
__device__ __forceinline__ v16bf load_bT_frag(const __bf16* src, int ld) {
  int lane = threadIdx.x & 31;
  int n = lane & 15, h = lane >> 4;
  v16bf f;
#pragma unroll
  for (int v = 0; v < 8; ++v)
#pragma unroll
    for (int e = 0; e < 2; ++e)
      f[v * 2 + e] = src[n * ld + a_kidx(v, e, h)];
  return f;
}

__device__ __forceinline__ v8f wmma_bf16(v16bf a, v16bf b, v8f c) {
  return __builtin_amdgcn_wmma_f32_16x16x32_bf16(false, a, false, b, (short)0, c,
                                                 false, false);
}

// ---------- kernel 0: weight repack + attn_w bf16 ----------
__global__ void pack_kernel(const float* __restrict__ w1, const float* __restrict__ w2,
                            const float* __restrict__ aw,
                            __bf16* __restrict__ wp1, __bf16* __restrict__ wp2,
                            __bf16* __restrict__ awb) {
  const int n1 = PPD * PPD * KW;
  for (int i = blockIdx.x * blockDim.x + threadIdx.x; i < n1;
       i += gridDim.x * blockDim.x) {
    int k = i % KW;
    int pi = (i / KW) % PPD;
    int po = i / (KW * PPD);
    int o = (k * PPD + po) * PPD + pi;   // [k][po][pi]
    wp1[o] = f2bf(w1[i]);
    wp2[o] = f2bf(w2[i]);
  }
  const int na = SS * LDIM;
  for (int i = blockIdx.x * blockDim.x + threadIdx.x; i < na;
       i += gridDim.x * blockDim.x)
    awb[i] = f2bf(aw[i]);
}

__global__ void zero_kernel(float* __restrict__ out, int n) {
  for (int i = blockIdx.x * blockDim.x + threadIdx.x; i < n;
       i += gridDim.x * blockDim.x)
    out[i] = 0.0f;
}

// ---------- kernel 1: projection GEMM + bias + validity mask -> h bf16 [B,S,P] ----------
__global__ __launch_bounds__(256) void proj_kernel(const float* __restrict__ x,
                                                   const float* __restrict__ pw,
                                                   const float* __restrict__ pb,
                                                   const int* __restrict__ seqlen,
                                                   __bf16* __restrict__ hbf) {
  __shared__ alignas(16) __bf16 As[16][40];    // [m][k]
  __shared__ alignas(16) __bf16 Bs[128][40];   // [n][k]  (transposed)
  const int tid = threadIdx.x, wave = tid >> 5, lane = tid & 31;
  const int m0 = blockIdx.y * 16;    // row = b*S + s
  const int n0 = blockIdx.x * 128;   // p
  v8f acc = {0.f, 0.f, 0.f, 0.f, 0.f, 0.f, 0.f, 0.f};
  for (int e0 = 0; e0 < EMBD; e0 += 32) {
    __syncthreads();
    for (int idx = tid; idx < 16 * 32; idx += 256) {
      int r = idx >> 5, kk = idx & 31;
      int row = m0 + r, b = row >> 10, s = row & (SS - 1);
      As[r][kk] = f2bf(x[((size_t)b * SS + s) * EMBD + e0 + kk]);
    }
    for (int idx = tid; idx < 128 * 32; idx += 256) {
      int n = idx >> 5, kk = idx & 31;  // coalesced: consecutive tid -> consecutive e
      Bs[n][kk] = f2bf(pw[(size_t)(n0 + n) * EMBD + e0 + kk]);
    }
    if (e0 + 32 < EMBD)
      __builtin_prefetch(&pw[(size_t)n0 * EMBD + e0 + 32], 0, 1);
    __syncthreads();
    v16bf a = load_a_frag(&As[0][0], 40);
    v16bf b = load_bT_frag(&Bs[wave * 16][0], 40);
    acc = wmma_bf16(a, b, acc);
  }
  const int nn = n0 + wave * 16 + (lane & 15);
  const int hf = lane >> 4;
  const float bias = pb[nn];
#pragma unroll
  for (int r = 0; r < 8; ++r) {
    int m = r + 8 * hf;
    int row = m0 + m, b = row >> 10, s = row & (SS - 1);
    float v = acc[r] + bias;
    if (!(s >= 1 && s <= seqlen[b])) v = 0.0f;
    hbf[((size_t)b * SS + s) * PPD + nn] = f2bf(v);
  }
}

// ---------- kernel 2: dual conv1d (implicit GEMM) + ReLU + add -> c bf16 [B,H,S,L] ----------
__global__ __launch_bounds__(256) void conv_kernel(const __bf16* __restrict__ hbf,
                                                   const __bf16* __restrict__ wp1,
                                                   const __bf16* __restrict__ wp2,
                                                   const float* __restrict__ b1,
                                                   const float* __restrict__ b2,
                                                   __bf16* __restrict__ cbf) {
  __shared__ alignas(16) __bf16 Hs[136][40];     // [s0-4 .. s0+131][pi]  (transposed)
  __shared__ alignas(16) __bf16 A1s[KW][16][32]; // [k][po][pi]
  __shared__ alignas(16) __bf16 A2s[KW][16][32];
  const int tid = threadIdx.x, wave = tid >> 5, lane = tid & 31;
  const int b = blockIdx.z;
  const int po0 = blockIdx.y * 16;
  const int s0 = blockIdx.x * 128;
  v8f acc1 = {0.f, 0.f, 0.f, 0.f, 0.f, 0.f, 0.f, 0.f};
  v8f acc2 = {0.f, 0.f, 0.f, 0.f, 0.f, 0.f, 0.f, 0.f};
  for (int pi0 = 0; pi0 < PPD; pi0 += 32) {
    __syncthreads();
    for (int idx = tid; idx < 136 * 32; idx += 256) {
      int j = idx >> 5, kk = idx & 31;  // coalesced: consecutive tid -> consecutive pi
      int si = s0 - 4 + j;
      __bf16 v = bf_zero();
      if (si >= 0 && si < SS) v = hbf[((size_t)b * SS + si) * PPD + pi0 + kk];
      Hs[j][kk] = v;
    }
    for (int idx = tid; idx < KW * 16 * 32; idx += 256) {
      int kk = idx & 31, m = (idx >> 5) & 15, k = idx >> 9;
      int src = (k * PPD + po0 + m) * PPD + pi0 + kk;
      A1s[k][m][kk] = wp1[src];
      A2s[k][m][kk] = wp2[src];
    }
    __syncthreads();
#pragma unroll
    for (int k = 0; k < KW; ++k) {
      v16bf bf = load_bT_frag(&Hs[wave * 16 + k][0], 40);
      v16bf a1 = load_a_frag(&A1s[k][0][0], 32);
      acc1 = wmma_bf16(a1, bf, acc1);
      v16bf a2 = load_a_frag(&A2s[k][0][0], 32);
      acc2 = wmma_bf16(a2, bf, acc2);
    }
  }
  const int s = s0 + wave * 16 + (lane & 15);
  const int hf = lane >> 4;
#pragma unroll
  for (int r = 0; r < 8; ++r) {
    int po = po0 + r + 8 * hf;
    float v1 = acc1[r] + b1[po]; v1 = v1 > 0.f ? v1 : 0.f;
    float v2 = acc2[r] + b2[po]; v2 = v2 > 0.f ? v2 : 0.f;
    int hh = po & (HEADS - 1);   // po = ll*H + hh
    int ll = po >> 3;
    cbf[(((size_t)b * HEADS + hh) * SS + s) * LDIM + ll] = f2bf(v1 + v2);
  }
}

// ---------- kernel 3: flash attention + masked sum over s -> rep[b,p] (atomic) ----------
__global__ __launch_bounds__(256) void attn_kernel(const __bf16* __restrict__ cbf,
                                                   const __bf16* __restrict__ awb,
                                                   const float* __restrict__ ab,
                                                   const int* __restrict__ seqlen,
                                                   float* __restrict__ rep) {
  __shared__ alignas(16) __bf16 Ws[32][72];     // attn_w tile [S'][l]  ([N][K] for score-B)
  __shared__ alignas(16) __bf16 Cst[64][40];    // c tile transposed [l][S'] ([N][K] for rep-B)
  __shared__ alignas(16) __bf16 Ps[8][16][40];  // per-wave P tile [s][S'chunk]
  const int tid = threadIdx.x, wave = tid >> 5, lane = tid & 31;
  const int b = blockIdx.z, hh = blockIdx.y;
  const int s0 = blockIdx.x * 128 + wave * 16;
  const int L = seqlen[b];
  const int half = lane >> 4, n16 = lane & 15;

  const __bf16* crow = cbf + (((size_t)b * HEADS + hh) * SS + s0) * LDIM;
  v16bf a0 = load_a_frag(crow, LDIM);        // l = 0..31
  v16bf a1 = load_a_frag(crow + 32, LDIM);   // l = 32..63

  float rmax[8], rsum[8];
#pragma unroll
  for (int r = 0; r < 8; ++r) { rmax[r] = -3.0e38f; rsum[r] = 0.f; }
  v8f o0 = {0.f,0.f,0.f,0.f,0.f,0.f,0.f,0.f};
  v8f o1 = o0, o2 = o0, o3 = o0;

  for (int t0 = 0; t0 < SS; t0 += 32) {
    __syncthreads();
    for (int idx = tid; idx < 32 * 64; idx += 256) {
      int sp = idx >> 6, l = idx & 63;  // coalesced reads (consecutive l)
      Ws[sp][l] = awb[(size_t)(t0 + sp) * LDIM + l];
      Cst[l][sp] = cbf[(((size_t)b * HEADS + hh) * SS + t0 + sp) * LDIM + l];
    }
    __syncthreads();
    // scores for 32 S' columns = two 16-col n-halves, K = 64 (two WMMAs each)
    v8f z = {0.f,0.f,0.f,0.f,0.f,0.f,0.f,0.f};
    v8f sc0 = wmma_bf16(a0, load_bT_frag(&Ws[0][0], 72), z);
    sc0 = wmma_bf16(a1, load_bT_frag(&Ws[0][32], 72), sc0);
    v8f sc1 = wmma_bf16(a0, load_bT_frag(&Ws[16][0], 72), z);
    sc1 = wmma_bf16(a1, load_bT_frag(&Ws[16][32], 72), sc1);

    const int sp0 = t0 + n16, sp1 = t0 + 16 + n16;
    const float bb0 = ab[sp0], bb1 = ab[sp1];
    const bool v0 = (sp0 >= 1 && sp0 <= L);
    const bool v1 = (sp1 >= 1 && sp1 <= L);
    float scale[8], p0[8], p1[8];
#pragma unroll
    for (int r = 0; r < 8; ++r) {
      float x0 = v0 ? sc0[r] + bb0 : -1.0e9f;
      float x1 = v1 ? sc1[r] + bb1 : -1.0e9f;
      float m = fmaxf(x0, x1);
      m = fmaxf(m, __shfl_xor(m, 1, 32));
      m = fmaxf(m, __shfl_xor(m, 2, 32));
      m = fmaxf(m, __shfl_xor(m, 4, 32));
      m = fmaxf(m, __shfl_xor(m, 8, 32));
      float nm = fmaxf(rmax[r], m);
      scale[r] = __expf(rmax[r] - nm);
      rmax[r] = nm;
      float e0 = __expf(x0 - nm), e1 = __expf(x1 - nm);
      float sr = e0 + e1;
      sr += __shfl_xor(sr, 1, 32);
      sr += __shfl_xor(sr, 2, 32);
      sr += __shfl_xor(sr, 4, 32);
      sr += __shfl_xor(sr, 8, 32);
      rsum[r] = rsum[r] * scale[r] + sr;
      p0[r] = e0; p1[r] = e1;
    }
    // D-layout -> A-fragment via wave-private LDS round trip
#pragma unroll
    for (int r = 0; r < 8; ++r) {
      Ps[wave][r + 8 * half][n16] = f2bf(p0[r]);
      Ps[wave][r + 8 * half][16 + n16] = f2bf(p1[r]);
    }
    asm volatile("s_wait_dscnt 0" ::: "memory");
    v16bf pa = load_a_frag(&Ps[wave][0][0], 40);
#pragma unroll
    for (int r = 0; r < 8; ++r) {
      o0[r] *= scale[r]; o1[r] *= scale[r];
      o2[r] *= scale[r]; o3[r] *= scale[r];
    }
    o0 = wmma_bf16(pa, load_bT_frag(&Cst[0][0],  40), o0);
    o1 = wmma_bf16(pa, load_bT_frag(&Cst[16][0], 40), o1);
    o2 = wmma_bf16(pa, load_bT_frag(&Cst[32][0], 40), o2);
    o3 = wmma_bf16(pa, load_bT_frag(&Cst[48][0], 40), o3);
  }
  // finalize softmax denom + valid(s) mask; column-sum over the 16 rows; atomically
  // accumulate into rep[b, p] (reference sums rep over s)
  float fac[8];
#pragma unroll
  for (int r = 0; r < 8; ++r) {
    int s = s0 + r + 8 * half;
    fac[r] = (s >= 1 && s <= L) ? 1.0f / rsum[r] : 0.0f;
  }
  float c0 = 0.f, c1 = 0.f, c2 = 0.f, c3 = 0.f;
#pragma unroll
  for (int r = 0; r < 8; ++r) {
    c0 += o0[r] * fac[r]; c1 += o1[r] * fac[r];
    c2 += o2[r] * fac[r]; c3 += o3[r] * fac[r];
  }
  c0 += __shfl_xor(c0, 16, 32);
  c1 += __shfl_xor(c1, 16, 32);
  c2 += __shfl_xor(c2, 16, 32);
  c3 += __shfl_xor(c3, 16, 32);
  if (lane < 16) {
    atomicAdd(&rep[b * PPD + ((0 + n16) * HEADS + hh)], c0);
    atomicAdd(&rep[b * PPD + ((16 + n16) * HEADS + hh)], c1);
    atomicAdd(&rep[b * PPD + ((32 + n16) * HEADS + hh)], c2);
    atomicAdd(&rep[b * PPD + ((48 + n16) * HEADS + hh)], c3);
  }
}

// ---------- kernel 4: classifier MLP + softmax + argmax ----------
__global__ __launch_bounds__(256) void cls_kernel(
    const float* __restrict__ rep,
    const float* w0, const float* w1, const float* w2, const float* w3,
    const float* w4, const float* w5, const float* w6, const float* w7,
    const float* c0, const float* c1, const float* c2, const float* c3,
    const float* c4, const float* c5, const float* c6, const float* c7,
    float* __restrict__ out) {
  __shared__ float zA[16][512];
  __shared__ float zB[16][256];
  const int tid = threadIdx.x;
  for (int i = tid; i < 16 * 512; i += 256) zA[i >> 9][i & 511] = rep[i];
  __syncthreads();
  const float* ws[8] = {w0, w1, w2, w3, w4, w5, w6, w7};
  const float* bs[8] = {c0, c1, c2, c3, c4, c5, c6, c7};
  const int dims[9] = {512, 256, 128, 64, 32, 16, 8, 4, 2};
  for (int layer = 0; layer < 8; ++layer) {
    int din = dims[layer], dout = dims[layer + 1];
    for (int idx = tid; idx < 16 * dout; idx += 256) {
      int bb = idx / dout, o = idx % dout;
      const float* wrow = ws[layer] + (size_t)o * din;
      float acc = bs[layer][o];
      if (layer & 1) {
        for (int i = 0; i < din; ++i) acc += wrow[i] * zB[bb][i];
      } else {
        for (int i = 0; i < din; ++i) acc += wrow[i] * zA[bb][i];
      }
      if (layer < 7) acc = acc > 0.f ? acc : 0.01f * acc;  // leaky relu
      if (layer & 1) zA[bb][o] = acc; else zB[bb][o] = acc;
    }
    __syncthreads();
  }
  if (tid < 16) {  // final logits in zA (layer 7 writes zA)
    float z0 = zA[tid][0], z1 = zA[tid][1];
    float m = fmaxf(z0, z1);
    float e0 = __expf(z0 - m), e1 = __expf(z1 - m);
    float s = e0 + e1;
    out[tid * 2 + 0] = e0 / s;
    out[tid * 2 + 1] = e1 / s;
    out[32 + tid] = (z1 > z0) ? 1.0f : 0.0f;
  }
}

extern "C" void kernel_launch(void* const* d_in, const int* in_sizes, int n_in,
                              void* d_out, int out_size, void* d_ws, size_t ws_size,
                              hipStream_t stream) {
  (void)in_sizes; (void)n_in; (void)ws_size;
  const float* x       = (const float*)d_in[0];
  const int*   seqlen  = (const int*)d_in[1];
  const float* proj_w  = (const float*)d_in[2];
  const float* proj_b  = (const float*)d_in[3];
  const float* conv1_w = (const float*)d_in[4];
  const float* conv1_b = (const float*)d_in[5];
  const float* conv2_w = (const float*)d_in[6];
  const float* conv2_b = (const float*)d_in[7];
  const float* attn_w  = (const float*)d_in[8];
  const float* attn_b  = (const float*)d_in[9];

  char* ws = (char*)d_ws;
  __bf16* hbf = (__bf16*)(ws + 0);                       // 16 MiB: [B,S,P] bf16
  __bf16* cbf = (__bf16*)(ws + (size_t)16777216);        // 16 MiB: [B,H,S,L] bf16
  __bf16* wp1 = (__bf16*)(ws + (size_t)33554432);        // 4.5 MiB: [k][po][pi]
  __bf16* wp2 = (__bf16*)(ws + (size_t)38273024);
  __bf16* awb = (__bf16*)(ws + (size_t)42991616);        // 128 KiB: attn_w bf16

  float* out = (float*)d_out;
  float* rep = out + 48;  // prob[32] + argmax[16] + rep[16*512]

  pack_kernel<<<2048, 256, 0, stream>>>(conv1_w, conv2_w, attn_w, wp1, wp2, awb);
  zero_kernel<<<64, 256, 0, stream>>>(out, out_size);
  proj_kernel<<<dim3(PPD / 128, (BB * SS) / 16), 256, 0, stream>>>(
      x, proj_w, proj_b, seqlen, hbf);
  conv_kernel<<<dim3(SS / 128, PPD / 16, BB), 256, 0, stream>>>(
      hbf, wp1, wp2, conv1_b, conv2_b, cbf);
  attn_kernel<<<dim3(SS / 128, HEADS, BB), 256, 0, stream>>>(
      cbf, awb, attn_b, seqlen, rep);
  cls_kernel<<<1, 256, 0, stream>>>(rep,
      (const float*)d_in[10], (const float*)d_in[11], (const float*)d_in[12],
      (const float*)d_in[13], (const float*)d_in[14], (const float*)d_in[15],
      (const float*)d_in[16], (const float*)d_in[17],
      (const float*)d_in[18], (const float*)d_in[19], (const float*)d_in[20],
      (const float*)d_in[21], (const float*)d_in[22], (const float*)d_in[23],
      (const float*)d_in[24], (const float*)d_in[25],
      out);
}
